// GeneralConv_10239202034065
// MI455X (gfx1250) — compile-verified, matrix operand order checked by reading, not run on hardware
//
#include <hip/hip_runtime.h>

#define N_NODES 50000
#define N_EDGES 800000

typedef __attribute__((ext_vector_type(16))) __bf16 v16bf;
typedef __attribute__((ext_vector_type(8)))  __bf16 v8bf;
typedef __attribute__((ext_vector_type(8)))  float  v8f;

// Fast activation helpers: v_exp_f32 + v_rcp_f32 / v_rsq_f32.
__device__ __forceinline__ float sigmoidf_(float x) {
  return __builtin_amdgcn_rcpf(1.0f + __expf(-x));
}
__device__ __forceinline__ float siluf_(float x) { return x * sigmoidf_(x); }

__device__ __forceinline__ v8f wmma_bf16(v16bf a, v16bf b, v8f c) {
  // D = A(16x32 bf16) * B(32x16 bf16) + C(16x16 f32)
  return __builtin_amdgcn_wmma_f32_16x16x32_bf16(false, a, false, b, (short)0, c,
                                                 false, false);
}

// ---------------------------------------------------------------------------
// One GEMM layer slab: acc[nc] = bias + sum_kc A[kc] * W[kc][nc].
// Software-pipelined, double-buffered LDS weight staging: the global loads for
// chunk kc+1 are issued BEFORE the 8-WMMA chain for chunk kc and only consumed
// (ds_store + s_wait_loadcnt) AFTER it, hiding the L2 latency under the math.
// Fully unrolled so A fragments stay in VGPRs (no scratch).
// ---------------------------------------------------------------------------
template <int KC>
__device__ __forceinline__ void gemm_layer(const v16bf* __restrict__ af,
                                           const __bf16* __restrict__ W,
                                           const float* __restrict__ bias,
                                           __bf16* __restrict__ wst0,
                                           __bf16* __restrict__ wst1,
                                           int tid, int lane, int col,
                                           v8f acc[8]) {
#pragma unroll
  for (int nc = 0; nc < 8; ++nc) {
    float bv = bias[nc * 16 + col];
#pragma unroll
    for (int r = 0; r < 8; ++r) acc[nc][r] = bv;
  }
  {  // prologue: stage chunk 0
    const v16bf* gs = (const v16bf*)W;
    v16bf* ld = (v16bf*)wst0;
    ld[tid]       = gs[tid];
    ld[tid + 128] = gs[tid + 128];
  }
  __syncthreads();
#pragma unroll
  for (int kc = 0; kc < KC; ++kc) {
    const __bf16* cur = (kc & 1) ? wst1 : wst0;
    v16bf s0, s1;
    if (kc + 1 < KC) {  // issue next-chunk loads early (no wait yet)
      const v16bf* gs = (const v16bf*)(W + (size_t)(kc + 1) * 4096);
      s0 = gs[tid];
      s1 = gs[tid + 128];
    }
#pragma unroll
    for (int nc = 0; nc < 8; ++nc) {
      v16bf b = *(const v16bf*)(cur + (nc * 32 + lane) * 16);
      acc[nc] = wmma_bf16(af[kc], b, acc[nc]);
    }
    if (kc + 1 < KC) {  // drain into the other buffer after the math
      v16bf* ld = (v16bf*)(((kc + 1) & 1) ? wst1 : wst0);
      ld[tid]       = s0;
      ld[tid + 128] = s1;
    }
    __syncthreads();
  }
}

// ---------------------------------------------------------------------------
// Weight swizzle: f32 [K x 128] row-major -> bf16 WMMA-B fragments.
// Fragment (kc, nc): 32 lanes x 16 bf16, contiguous per lane (32B vector load).
// Lane layout (ISA 7.12.2, 16-bit B 32x16): lane<16 holds col N=lane,
//   elems 0..7 = K k0..k0+7, elems 8..15 = K k0+16..k0+23; lane>=16 offsets K by 8.
// ---------------------------------------------------------------------------
__global__ void swizzle_w(const float* __restrict__ W, int Kreal, int total,
                          __bf16* __restrict__ dst) {
  int idx = blockIdx.x * 256 + threadIdx.x;
  if (idx >= total) return;
  int t    = idx & 15;
  int lane = (idx >> 4) & 31;
  int nc   = (idx >> 9) & 7;
  int kc   = idx >> 12;
  int n    = nc * 16 + (lane & 15);
  int off  = (lane >> 4) * 8;
  int k    = kc * 32 + ((t < 8) ? (off + t) : (16 + off + (t - 8)));
  float v  = (k < Kreal) ? W[(size_t)k * 128 + n] : 0.0f;
  dst[idx] = (__bf16)v;
}

// Edge-count histogram for scatter-mean.
__global__ void count_kernel(const int* __restrict__ edge_index,
                             float* __restrict__ counts) {
  int e = blockIdx.x * 256 + threadIdx.x;
  if (e < N_EDGES) atomicAdd(&counts[edge_index[e]], 1.0f);
}

// ---------------------------------------------------------------------------
// Edge GatedMLP: x_e=[ninp[src]|ninp[dst]|efeat] (384) -> silu(h=128) -> 128,
// gate branch likewise -> out = m * sigmoid(g); atomic scatter-add into sums.
// One wave per 16 edges; 4 waves share double-buffered staged weights.
// ---------------------------------------------------------------------------
__global__ __launch_bounds__(128, 1) void edge_mlp_kernel(
    const float* __restrict__ node_inp, const float* __restrict__ edge_feature,
    const int* __restrict__ edge_index,
    const float* __restrict__ bm0, const float* __restrict__ bm1,
    const float* __restrict__ bg0, const float* __restrict__ bg1,
    const __bf16* __restrict__ Wm0s, const __bf16* __restrict__ Wm1s,
    const __bf16* __restrict__ Wg0s, const __bf16* __restrict__ Wg1s,
    float* __restrict__ sums) {
  __shared__ __align__(32) __bf16 wst0[4096];          // 8 KB stage buf 0
  __shared__ __align__(32) __bf16 wst1[4096];          // 8 KB stage buf 1
  __shared__ __align__(16) __bf16 hidbuf[4][2][2048];  // 32 KB hidden
  const int tid  = threadIdx.x;
  const int wave = tid >> 5;
  const int lane = tid & 31;
  const int hi   = lane >> 4;
  const int col  = lane & 15;
  const int eBase = (blockIdx.x * 4 + wave) * 16;

  const int* srcIdx = edge_index;
  const int* dstIdx = edge_index + N_EDGES;

  const int eRow = eBase + col;  // this lane's A-matrix row (M = col)
  const int sN = srcIdx[eRow];
  const int dN = dstIdx[eRow];

  // ---- Build 12 A fragments (K = 384 = 3 x 128), f32 -> bf16 on the fly ----
  v16bf af[12];
  {
    const float* p0 = node_inp + (size_t)sN * 128;
    const float* p1 = node_inp + (size_t)dN * 128;
    const float* p2 = edge_feature + (size_t)eRow * 128;
#pragma unroll
    for (int kc = 0; kc < 4; ++kc) {
      int k0 = kc * 32 + hi * 8;
      v16bf a0, a1, a2;
#pragma unroll
      for (int t = 0; t < 8; ++t) {
        a0[t] = (__bf16)p0[k0 + t];  a0[8 + t] = (__bf16)p0[k0 + 16 + t];
        a1[t] = (__bf16)p1[k0 + t];  a1[8 + t] = (__bf16)p1[k0 + 16 + t];
        a2[t] = (__bf16)p2[k0 + t];  a2[8 + t] = (__bf16)p2[k0 + 16 + t];
      }
      af[kc] = a0; af[4 + kc] = a1; af[8 + kc] = a2;
    }
  }

  // ---- Layer 0 (both branches): hidden = silu(x @ W0 + b0) -> LDS bf16 ----
  __bf16* hm = &hidbuf[wave][0][0];
  __bf16* hg = &hidbuf[wave][1][0];
  {
    v8f acc[8];
    gemm_layer<12>(af, Wm0s, bm0, wst0, wst1, tid, lane, col, acc);
#pragma unroll
    for (int nc = 0; nc < 8; ++nc)
#pragma unroll
      for (int r = 0; r < 8; ++r)
        hm[(r + hi * 8) * 128 + nc * 16 + col] = (__bf16)siluf_(acc[nc][r]);

    gemm_layer<12>(af, Wg0s, bg0, wst0, wst1, tid, lane, col, acc);
#pragma unroll
    for (int nc = 0; nc < 8; ++nc)
#pragma unroll
      for (int r = 0; r < 8; ++r)
        hg[(r + hi * 8) * 128 + nc * 16 + col] = (__bf16)siluf_(acc[nc][r]);
  }
  asm volatile("s_wait_dscnt 0x0" ::: "memory");

  // ---- Re-load hidden as A fragments (K = 128 -> 4 chunks) ----
  v16bf am[4], ag[4];
  {
    const __bf16* hrm = hm + col * 128;
    const __bf16* hrg = hg + col * 128;
#pragma unroll
    for (int kc = 0; kc < 4; ++kc) {
      int k0 = kc * 32 + hi * 8;
      v8bf l0 = *(const v8bf*)(hrm + k0);
      v8bf l1 = *(const v8bf*)(hrm + k0 + 16);
      am[kc] = __builtin_shufflevector(l0, l1, 0,1,2,3,4,5,6,7,8,9,10,11,12,13,14,15);
      v8bf g0 = *(const v8bf*)(hrg + k0);
      v8bf g1 = *(const v8bf*)(hrg + k0 + 16);
      ag[kc] = __builtin_shufflevector(g0, g1, 0,1,2,3,4,5,6,7,8,9,10,11,12,13,14,15);
    }
  }
  asm volatile("s_wait_dscnt 0x0" ::: "memory");

  // src node for each of this lane's 8 output rows (D-frag: M = r + 8*hi)
  int srows[8];
#pragma unroll
  for (int r = 0; r < 8; ++r) srows[r] = srcIdx[eBase + r + hi * 8];

  // ---- Layer 1: park main branch in LDS f32, then gate + atomic scatter ----
  float* outm = (float*)&hidbuf[wave][0][0];  // 16x128 f32 reuses the 8 KB slice
  {
    v8f acc[8];
    gemm_layer<4>(am, Wm1s, bm1, wst0, wst1, tid, lane, col, acc);
#pragma unroll
    for (int nc = 0; nc < 8; ++nc)
#pragma unroll
      for (int r = 0; r < 8; ++r)
        outm[(r + hi * 8) * 128 + nc * 16 + col] = acc[nc][r];

    gemm_layer<4>(ag, Wg1s, bg1, wst0, wst1, tid, lane, col, acc);
    asm volatile("s_wait_dscnt 0x0" ::: "memory");
#pragma unroll
    for (int nc = 0; nc < 8; ++nc) {
      int n = nc * 16 + col;
#pragma unroll
      for (int r = 0; r < 8; ++r) {
        float v = outm[(r + hi * 8) * 128 + n] * sigmoidf_(acc[nc][r]);
        atomicAdd(&sums[(size_t)srows[r] * 128 + n], v);
      }
    }
  }
}

// ---------------------------------------------------------------------------
// Node GatedMLP: x_n=[ninp|agg|coords|gs] (385, padded to 416 = 13 K-chunks)
// -> GatedMLP -> LayerNorm. One wave per 16 nodes; 4 waves share staged weights.
// ---------------------------------------------------------------------------
__global__ __launch_bounds__(128, 1) void node_mlp_kernel(
    const float* __restrict__ node_inp, const float* __restrict__ coords,
    const float* __restrict__ gstate, const float* __restrict__ counts,
    const float* __restrict__ sums,
    const float* __restrict__ bm0, const float* __restrict__ bm1,
    const float* __restrict__ bg0, const float* __restrict__ bg1,
    const __bf16* __restrict__ Wm0s, const __bf16* __restrict__ Wm1s,
    const __bf16* __restrict__ Wg0s, const __bf16* __restrict__ Wg1s,
    const float* __restrict__ gamma, const float* __restrict__ beta,
    float* __restrict__ out) {
  __shared__ __align__(32) __bf16 wst0[4096];          // 8 KB stage buf 0
  __shared__ __align__(32) __bf16 wst1[4096];          // 8 KB stage buf 1
  __shared__ __align__(16) __bf16 hidbuf[4][2][2048];  // 32 KB hidden
  const int tid  = threadIdx.x;
  const int wave = tid >> 5;
  const int lane = tid & 31;
  const int hi   = lane >> 4;
  const int col  = lane & 15;
  const int nodeBase = (blockIdx.x * 4 + wave) * 16;

  const int rowNode = nodeBase + col;
  const int nClamp  = rowNode < N_NODES ? rowNode : (N_NODES - 1);

  float cnt = counts[nClamp];
  float aggScale = cnt > 0.0f ? __builtin_amdgcn_rcpf(cnt) : 0.0f;  // mean fold

  // ---- A fragments: 12 real K-chunks + synthesized chunk 12 (gs | zeros) ----
  v16bf af[13];
  {
    const float* p0 = node_inp + (size_t)nClamp * 128;
    const float* p1 = sums + (size_t)nClamp * 128;
    const float* p2 = coords + (size_t)nClamp * 128;
#pragma unroll
    for (int kc = 0; kc < 4; ++kc) {
      int k0 = kc * 32 + hi * 8;
      v16bf a0, a1, a2;
#pragma unroll
      for (int t = 0; t < 8; ++t) {
        a0[t] = (__bf16)p0[k0 + t];               a0[8 + t] = (__bf16)p0[k0 + 16 + t];
        a1[t] = (__bf16)(p1[k0 + t] * aggScale);  a1[8 + t] = (__bf16)(p1[k0 + 16 + t] * aggScale);
        a2[t] = (__bf16)p2[k0 + t];               a2[8 + t] = (__bf16)p2[k0 + 16 + t];
      }
      af[kc] = a0; af[4 + kc] = a1; af[8 + kc] = a2;
    }
    v16bf a12;
#pragma unroll
    for (int t = 0; t < 16; ++t) a12[t] = (__bf16)0.0f;
    if (hi == 0) a12[0] = (__bf16)gstate[nClamp];  // K=384 column = global_state
    af[12] = a12;
  }

  __bf16* hm = &hidbuf[wave][0][0];
  __bf16* hg = &hidbuf[wave][1][0];
  {
    v8f acc[8];
    gemm_layer<13>(af, Wm0s, bm0, wst0, wst1, tid, lane, col, acc);
#pragma unroll
    for (int nc = 0; nc < 8; ++nc)
#pragma unroll
      for (int r = 0; r < 8; ++r)
        hm[(r + hi * 8) * 128 + nc * 16 + col] = (__bf16)siluf_(acc[nc][r]);

    gemm_layer<13>(af, Wg0s, bg0, wst0, wst1, tid, lane, col, acc);
#pragma unroll
    for (int nc = 0; nc < 8; ++nc)
#pragma unroll
      for (int r = 0; r < 8; ++r)
        hg[(r + hi * 8) * 128 + nc * 16 + col] = (__bf16)siluf_(acc[nc][r]);
  }
  asm volatile("s_wait_dscnt 0x0" ::: "memory");

  v16bf am[4], ag[4];
  {
    const __bf16* hrm = hm + col * 128;
    const __bf16* hrg = hg + col * 128;
#pragma unroll
    for (int kc = 0; kc < 4; ++kc) {
      int k0 = kc * 32 + hi * 8;
      v8bf l0 = *(const v8bf*)(hrm + k0);
      v8bf l1 = *(const v8bf*)(hrm + k0 + 16);
      am[kc] = __builtin_shufflevector(l0, l1, 0,1,2,3,4,5,6,7,8,9,10,11,12,13,14,15);
      v8bf g0 = *(const v8bf*)(hrg + k0);
      v8bf g1 = *(const v8bf*)(hrg + k0 + 16);
      ag[kc] = __builtin_shufflevector(g0, g1, 0,1,2,3,4,5,6,7,8,9,10,11,12,13,14,15);
    }
  }
  asm volatile("s_wait_dscnt 0x0" ::: "memory");

  // ---- Layer 1: main branch parked in LDS f32, gate applied in place -------
  float* outm = (float*)&hidbuf[wave][0][0];  // 16x128 f32 reuses the 8 KB slice
  {
    v8f acc[8];
    gemm_layer<4>(am, Wm1s, bm1, wst0, wst1, tid, lane, col, acc);
#pragma unroll
    for (int nc = 0; nc < 8; ++nc)
#pragma unroll
      for (int r = 0; r < 8; ++r)
        outm[(r + hi * 8) * 128 + nc * 16 + col] = acc[nc][r];

    gemm_layer<4>(ag, Wg1s, bg1, wst0, wst1, tid, lane, col, acc);
    asm volatile("s_wait_dscnt 0x0" ::: "memory");
#pragma unroll
    for (int nc = 0; nc < 8; ++nc) {
      int n = nc * 16 + col;
#pragma unroll
      for (int r = 0; r < 8; ++r) {
        int idx = (r + hi * 8) * 128 + n;
        outm[idx] = outm[idx] * sigmoidf_(acc[nc][r]);
      }
    }
  }
  asm volatile("s_wait_dscnt 0x0" ::: "memory");

  // ---- LayerNorm, one row per lane (lanes 0..15) ----
  if (lane < 16) {
    int node = nodeBase + lane;
    if (node < N_NODES) {
      const float* row = outm + lane * 128;
      float mu = 0.0f;
#pragma unroll
      for (int j = 0; j < 128; ++j) mu += row[j];
      mu *= (1.0f / 128.0f);
      float var = 0.0f;
#pragma unroll
      for (int j = 0; j < 128; ++j) { float d = row[j] - mu; var += d * d; }
      var *= (1.0f / 128.0f);
      float inv = __builtin_amdgcn_rsqf(var + 1e-5f);
      float* orow = out + (size_t)node * 128;
#pragma unroll
      for (int j = 0; j < 128; ++j)
        orow[j] = (row[j] - mu) * inv * gamma[j] + beta[j];
    }
  }
}

// ---------------------------------------------------------------------------
extern "C" void kernel_launch(void* const* d_in, const int* in_sizes, int n_in,
                              void* d_out, int out_size, void* d_ws, size_t ws_size,
                              hipStream_t stream) {
  const float* node_inp     = (const float*)d_in[0];
  const float* edge_feature = (const float*)d_in[1];
  const float* global_state = (const float*)d_in[2];
  /* d_in[3] = cells (unused by reference) */
  const float* coords       = (const float*)d_in[4];
  const int*   edge_index   = (const int*)d_in[5];
  const float* bond_Wm0 = (const float*)d_in[6];
  const float* bond_bm0 = (const float*)d_in[7];
  const float* bond_Wm1 = (const float*)d_in[8];
  const float* bond_bm1 = (const float*)d_in[9];
  const float* bond_Wg0 = (const float*)d_in[10];
  const float* bond_bg0 = (const float*)d_in[11];
  const float* bond_Wg1 = (const float*)d_in[12];
  const float* bond_bg1 = (const float*)d_in[13];
  const float* node_Wm0 = (const float*)d_in[14];
  const float* node_bm0 = (const float*)d_in[15];
  const float* node_Wm1 = (const float*)d_in[16];
  const float* node_bm1 = (const float*)d_in[17];
  const float* node_Wg0 = (const float*)d_in[18];
  const float* node_bg0 = (const float*)d_in[19];
  const float* node_Wg1 = (const float*)d_in[20];
  const float* node_bg1 = (const float*)d_in[21];
  const float* ln_gamma = (const float*)d_in[22];
  const float* ln_beta  = (const float*)d_in[23];
  float* out = (float*)d_out;

  // ---- workspace layout ----
  float*  sums   = (float*)d_ws;                  // N*128 f32
  float*  counts = sums + (size_t)N_NODES * 128;  // N f32
  __bf16* w      = (__bf16*)(counts + N_NODES);   // swizzled bf16 weights
  __bf16* bWm0 = w;                    // 12 KC
  __bf16* bWg0 = bWm0 + 12 * 4096;
  __bf16* bWm1 = bWg0 + 12 * 4096;     // 4 KC
  __bf16* bWg1 = bWm1 + 4 * 4096;
  __bf16* nWm0 = bWg1 + 4 * 4096;      // 13 KC (padded 385 -> 416)
  __bf16* nWg0 = nWm0 + 13 * 4096;
  __bf16* nWm1 = nWg0 + 13 * 4096;     // 4 KC
  __bf16* nWg1 = nWm1 + 4 * 4096;

  hipMemsetAsync(d_ws, 0, ((size_t)N_NODES * 128 + N_NODES) * sizeof(float), stream);

  auto swz = [&](const float* W, int Kreal, int KC, __bf16* dst) {
    int total = KC << 12;  // KC * 8 nc * 32 lanes * 16 elems
    swizzle_w<<<(total + 255) / 256, 256, 0, stream>>>(W, Kreal, total, dst);
  };
  swz(bond_Wm0, 384, 12, bWm0);
  swz(bond_Wg0, 384, 12, bWg0);
  swz(bond_Wm1, 128, 4,  bWm1);
  swz(bond_Wg1, 128, 4,  bWg1);
  swz(node_Wm0, 385, 13, nWm0);
  swz(node_Wg0, 385, 13, nWg0);
  swz(node_Wm1, 128, 4,  nWm1);
  swz(node_Wg1, 128, 4,  nWg1);

  count_kernel<<<(N_EDGES + 255) / 256, 256, 0, stream>>>(edge_index, counts);

  edge_mlp_kernel<<<N_EDGES / 64, 128, 0, stream>>>(
      node_inp, edge_feature, edge_index,
      bond_bm0, bond_bm1, bond_bg0, bond_bg1,
      bWm0, bWm1, bWg0, bWg1, sums);

  node_mlp_kernel<<<(N_NODES + 63) / 64, 128, 0, stream>>>(
      node_inp, coords, global_state, counts, sums,
      node_bm0, node_bm1, node_bg0, node_bg1,
      nWm0, nWm1, nWg0, nWg1, ln_gamma, ln_beta, out);
}
